// GIN_70188355551832
// MI455X (gfx1250) — matrix operand order repaired and checked by reading, not compiled, not run on hardware
//
#include <hip/hip_runtime.h>
#include <hip/hip_bf16.h>

// ---------------------------------------------------------------------------
// GIN (3-layer) forward on gfx1250.
// Edge aggregation: wave/edge, float4 gather + global_atomic_add_f32 scatter.
// GEMMs: v_wmma_f32_16x16x32_bf16, one wave -> 16x64 output strip.
// ---------------------------------------------------------------------------

typedef __attribute__((ext_vector_type(16))) __bf16 v16bf;
typedef __attribute__((ext_vector_type(8)))  __bf16 v8bf;
typedef __attribute__((ext_vector_type(8)))  float  v8f;

__device__ __forceinline__ __bf16 f2bf(float f) { return (__bf16)f; }

// ---------------- aggregation ----------------

// agg[i] = (1 + eps) * h[i]
__global__ void k_init_agg(const float* __restrict__ h, const float* __restrict__ epsp,
                           float* __restrict__ agg, int n) {
    int i = blockIdx.x * blockDim.x + threadIdx.x;
    if (i < n) {
        float e = 1.0f + epsp[0];
        agg[i] = e * h[i];
    }
}

// one wave per edge; lane handles 4 contiguous floats (32 lanes * 4 = 128 feats)
__global__ void k_scatter(const float* __restrict__ h, const int* __restrict__ src,
                          const int* __restrict__ dst, float* __restrict__ agg, int E) {
    int wid  = (blockIdx.x * blockDim.x + threadIdx.x) >> 5;
    int lane = threadIdx.x & 31;
    if (wid >= E) return;
    int s = src[wid];
    int d = dst[wid];
    const float4 v = ((const float4*)(h + (size_t)s * 128))[lane];
    float* ap = agg + (size_t)d * 128 + lane * 4;
    atomicAdd(ap + 0, v.x);
    atomicAdd(ap + 1, v.y);
    atomicAdd(ap + 2, v.z);
    atomicAdd(ap + 3, v.w);
}

// ---------------- conversions ----------------

__global__ void k_f2bf(const float* __restrict__ in, __bf16* __restrict__ out, int n) {
    int i = blockIdx.x * blockDim.x + threadIdx.x;
    if (i < n) out[i] = f2bf(in[i]);
}

// W [K, Nout] fp32 row-major  ->  Wt [Nout, K] bf16
__global__ void k_transpose_bf(const float* __restrict__ w, __bf16* __restrict__ wt,
                               int K, int Nout) {
    int i = blockIdx.x * blockDim.x + threadIdx.x;
    if (i < K * Nout) {
        int k = i / Nout, n = i % Nout;
        wt[(size_t)n * K + k] = f2bf(w[i]);
    }
}

__global__ void k_zero(float* __restrict__ p, int n) {
    int i = blockIdx.x * blockDim.x + threadIdx.x;
    if (i < n) p[i] = 0.0f;
}

// ---------------- batchnorm (training-mode stats) ----------------

// blockDim.x == C; each thread owns channel threadIdx.x for ROWS rows
__global__ void k_bn_stats(const float* __restrict__ h, float* __restrict__ sum,
                           float* __restrict__ sumsq, int N, int C, int rows) {
    int c  = threadIdx.x;
    int r0 = blockIdx.x * rows;
    float s = 0.0f, s2 = 0.0f;
    for (int r = 0; r < rows; ++r) {
        int row = r0 + r;
        if (row < N) {
            float v = h[(size_t)row * C + c];
            s  += v;
            s2 += v * v;
        }
    }
    atomicAdd(&sum[c], s);
    atomicAdd(&sumsq[c], s2);
}

__global__ void k_bn_final(const float* __restrict__ sum, const float* __restrict__ sumsq,
                           const float* __restrict__ g, const float* __restrict__ be,
                           float* __restrict__ scale, float* __restrict__ shift,
                           int N, int C) {
    int c = threadIdx.x;
    if (c < C) {
        float invN = 1.0f / (float)N;
        float mean = sum[c] * invN;
        float var  = sumsq[c] * invN - mean * mean;
        float sc   = g[c] * rsqrtf(var + 1e-5f);
        scale[c] = sc;
        shift[c] = be[c] - mean * sc;
    }
}

// out_bf16 = bf16(relu(scale*h + shift))
__global__ void k_bn_relu_bf16(const float* __restrict__ h, const float* __restrict__ sc,
                               const float* __restrict__ sh, __bf16* __restrict__ out,
                               int n, int C) {
    int i = blockIdx.x * blockDim.x + threadIdx.x;
    if (i < n) {
        int c = i % C;
        out[i] = f2bf(fmaxf(0.0f, fmaf(sc[c], h[i], sh[c])));
    }
}

// in-place fp32: h = relu(scale*h + shift)
__global__ void k_bn_relu_f32(float* __restrict__ h, const float* __restrict__ sc,
                              const float* __restrict__ sh, int n, int C) {
    int i = blockIdx.x * blockDim.x + threadIdx.x;
    if (i < n) {
        int c = i % C;
        h[i] = fmaxf(0.0f, fmaf(sc[c], h[i], sh[c]));
    }
}

// ---------------- WMMA GEMM ----------------
// C[M,Nout] = A[M,K](bf16) * B[K,Nout] + bias ; B supplied transposed: Bt[Nout,K].
// One wave computes a 16(M) x 64(N) strip: 4 accumulators, K stepped by 32.
// A fragment layout (16-bit A 16x32): lane<16 -> row=lane, K = kg .. kg+15 contiguous
// with kg = (lane>>4)*16.  B fragment (16-bit B 32x16): col=lane&15,
// K = (lane>>4)*16 .. +15 contiguous.  C/D: col=lane&15, rows r + 8*(lane>>4).
__global__ void k_gemm_bf16(const __bf16* __restrict__ A, const __bf16* __restrict__ Bt,
                            const float* __restrict__ bias, float* __restrict__ C,
                            int M, int K, int Nout) {
    const int ntiles = Nout >> 6;                     // 64-wide strips
    const int total  = (M >> 4) * ntiles;
    int wid = (blockIdx.x * blockDim.x + threadIdx.x) >> 5;
    if (wid >= total) return;
    int lane  = threadIdx.x & 31;
    int mtile = wid / ntiles;
    int ntile = wid % ntiles;

    const int l15 = lane & 15;
    const int hi  = lane >> 4;          // 0 or 1
    const int akg = hi * 16;            // A K-group base
    const int bkg = hi * 16;            // B K-group base
    const int nb  = ntile * 64;

    const __bf16* arow = A + (size_t)(mtile * 16 + l15) * K;
    const __bf16* b0   = Bt + (size_t)(nb + 0 * 16 + l15) * K + bkg;
    const __bf16* b1   = Bt + (size_t)(nb + 1 * 16 + l15) * K + bkg;
    const __bf16* b2   = Bt + (size_t)(nb + 2 * 16 + l15) * K + bkg;
    const __bf16* b3   = Bt + (size_t)(nb + 3 * 16 + l15) * K + bkg;

    v8f acc0 = {}, acc1 = {}, acc2 = {}, acc3 = {};

    for (int k0 = 0; k0 < K; k0 += 32) {
        v8bf alo = *(const v8bf*)(arow + k0 + akg);
        v8bf ahi = *(const v8bf*)(arow + k0 + akg + 8);
        v16bf a = __builtin_shufflevector(alo, ahi,
                      0, 1, 2, 3, 4, 5, 6, 7, 8, 9, 10, 11, 12, 13, 14, 15);
        v16bf bb0 = *(const v16bf*)(b0 + k0);
        v16bf bb1 = *(const v16bf*)(b1 + k0);
        v16bf bb2 = *(const v16bf*)(b2 + k0);
        v16bf bb3 = *(const v16bf*)(b3 + k0);
        acc0 = __builtin_amdgcn_wmma_f32_16x16x32_bf16(false, a, false, bb0, (short)0, acc0, false, false);
        acc1 = __builtin_amdgcn_wmma_f32_16x16x32_bf16(false, a, false, bb1, (short)0, acc1, false, false);
        acc2 = __builtin_amdgcn_wmma_f32_16x16x32_bf16(false, a, false, bb2, (short)0, acc2, false, false);
        acc3 = __builtin_amdgcn_wmma_f32_16x16x32_bf16(false, a, false, bb3, (short)0, acc3, false, false);
    }

    int rbase = mtile * 16 + 8 * hi;
    v8f accs[4] = {acc0, acc1, acc2, acc3};
#pragma unroll
    for (int sub = 0; sub < 4; ++sub) {
        int colg = nb + sub * 16 + l15;
        float bv = bias[colg];
#pragma unroll
        for (int r = 0; r < 8; ++r) {
            C[(size_t)(rbase + r) * Nout + colg] = accs[sub][r] + bv;
        }
    }
}

// ---------------- log-softmax (one wave per row of 128) ----------------

__global__ void k_logsoftmax(const float* __restrict__ h, float* __restrict__ out, int N) {
    int row  = (blockIdx.x * blockDim.x + threadIdx.x) >> 5;
    int lane = threadIdx.x & 31;
    if (row >= N) return;
    const float* r = h + (size_t)row * 128;
    float v0 = r[lane], v1 = r[lane + 32], v2 = r[lane + 64], v3 = r[lane + 96];
    float m = fmaxf(fmaxf(v0, v1), fmaxf(v2, v3));
    for (int off = 16; off > 0; off >>= 1) m = fmaxf(m, __shfl_xor(m, off, 32));
    float s = expf(v0 - m) + expf(v1 - m) + expf(v2 - m) + expf(v3 - m);
    for (int off = 16; off > 0; off >>= 1) s += __shfl_xor(s, off, 32);
    float l = m + logf(s);
    float* o = out + (size_t)row * 128;
    o[lane]      = v0 - l;
    o[lane + 32] = v1 - l;
    o[lane + 64] = v2 - l;
    o[lane + 96] = v3 - l;
}

// ---------------------------------------------------------------------------

extern "C" void kernel_launch(void* const* d_in, const int* in_sizes, int n_in,
                              void* d_out, int out_size, void* d_ws, size_t ws_size,
                              hipStream_t stream) {
    const int F   = 128;   // feature width (F_IN == H == 128)
    const int HID = 256;   // hidden width (2H)
    const int N = in_sizes[0] / F;
    const int E = in_sizes[1] / 2;

    const float* x   = (const float*)d_in[0];
    const int*   ei  = (const int*)d_in[1];
    const int*   src = ei;
    const int*   dst = ei + E;

    // per-layer params, insertion order: eps,W1,b1,g1,be1,W2,b2 (+bn_g,bn_b for i<2)
    const float *eps[3], *W1[3], *b1[3], *g1[3], *be1[3], *W2[3], *b2[3];
    const float *bng[2], *bnb[2];
    {
        int idx = 2;
        for (int i = 0; i < 3; ++i) {
            eps[i] = (const float*)d_in[idx + 0];
            W1[i]  = (const float*)d_in[idx + 1];
            b1[i]  = (const float*)d_in[idx + 2];
            g1[i]  = (const float*)d_in[idx + 3];
            be1[i] = (const float*)d_in[idx + 4];
            W2[i]  = (const float*)d_in[idx + 5];
            b2[i]  = (const float*)d_in[idx + 6];
            idx += 7;
            if (i < 2) {
                bng[i] = (const float*)d_in[idx + 0];
                bnb[i] = (const float*)d_in[idx + 1];
                idx += 2;
            }
        }
    }

    // ---- workspace carve-up (all sizes multiples of 256B) ----
    char* ws = (char*)d_ws;
    size_t o = 0;
    float*  agg   = (float*)(ws + o);  o += (size_t)N * F * 4;      // 25.6 MB
    float*  hmid  = (float*)(ws + o);  o += (size_t)N * HID * 4;    // 51.2 MB
    float*  hbuf  = (float*)(ws + o);  o += (size_t)N * F * 4;      // 25.6 MB (conv out / next h)
    __bf16* aggb  = (__bf16*)(ws + o); o += (size_t)N * F * 2;      // 12.8 MB
    __bf16* hmidb = (__bf16*)(ws + o); o += (size_t)N * HID * 2;    // 25.6 MB
    __bf16* w1t   = (__bf16*)(ws + o); o += (size_t)HID * F * 2;    // 64 KB  [256,128]
    __bf16* w2t   = (__bf16*)(ws + o); o += (size_t)F * HID * 2;    // 64 KB  [128,256]
    float*  ssum  = (float*)(ws + o);  o += 256 * 4;
    float*  ssq   = (float*)(ws + o);  o += 256 * 4;
    float*  scale = (float*)(ws + o);  o += 256 * 4;
    float*  shift = (float*)(ws + o);  o += 256 * 4;
    (void)ws_size; (void)n_in; (void)out_size;

    const int T = 256;
    const int ROWS = 64;                       // rows per BN-stats block
    float* outp = (float*)d_out;

    const float* h_cur = x;
    for (int i = 0; i < 3; ++i) {
        // ---- aggregation: agg = (1+eps)*h + segment_sum(h[src] -> dst) ----
        int nF = N * F;
        k_init_agg<<<(nF + T - 1) / T, T, 0, stream>>>(h_cur, eps[i], agg, nF);
        {
            long long thr = (long long)E * 32;
            int blocks = (int)((thr + T - 1) / T);
            k_scatter<<<blocks, T, 0, stream>>>(h_cur, src, dst, agg, E);
        }
        k_f2bf<<<(nF + T - 1) / T, T, 0, stream>>>(agg, aggb, nF);

        // ---- GEMM1: hmid[N,256] = agg @ W1 + b1 ----
        k_transpose_bf<<<(F * HID + T - 1) / T, T, 0, stream>>>(W1[i], w1t, F, HID);
        {
            int total = (N / 16) * (HID / 64);
            k_gemm_bf16<<<(total + 7) / 8, T, 0, stream>>>(aggb, w1t, b1[i], hmid, N, F, HID);
        }

        // ---- internal BN(g1,be1) + ReLU -> bf16 ----
        k_zero<<<1, 2 * HID, 0, stream>>>(ssum, 2 * HID);   // zeroes ssum and ssq (contiguous)
        k_bn_stats<<<(N + ROWS - 1) / ROWS, HID, 0, stream>>>(hmid, ssum, ssq, N, HID, ROWS);
        k_bn_final<<<1, HID, 0, stream>>>(ssum, ssq, g1[i], be1[i], scale, shift, N, HID);
        {
            int nH = N * HID;
            k_bn_relu_bf16<<<(nH + T - 1) / T, T, 0, stream>>>(hmid, scale, shift, hmidb, nH, HID);
        }

        // ---- GEMM2: hbuf[N,128] = hmid @ W2 + b2 ----
        k_transpose_bf<<<(HID * F + T - 1) / T, T, 0, stream>>>(W2[i], w2t, HID, F);
        {
            int total = (N / 16) * (F / 64);
            k_gemm_bf16<<<(total + 7) / 8, T, 0, stream>>>(hmidb, w2t, b2[i], hbuf, N, HID, F);
        }

        if (i < 2) {
            // ---- inter-layer BN + ReLU (fp32, in place) ----
            k_zero<<<1, 2 * F, 0, stream>>>(ssum, 2 * F);   // only first F of each used;
            k_zero<<<1, 2 * F, 0, stream>>>(ssq, 2 * F);    // zero both explicitly
            k_bn_stats<<<(N + ROWS - 1) / ROWS, F, 0, stream>>>(hbuf, ssum, ssq, N, F, ROWS);
            k_bn_final<<<1, F, 0, stream>>>(ssum, ssq, bng[i], bnb[i], scale, shift, N, F);
            k_bn_relu_f32<<<(nF + T - 1) / T, T, 0, stream>>>(hbuf, scale, shift, nF, F);
            h_cur = hbuf;
        } else {
            // ---- final log_softmax ----
            int blocks = (N * 32 + T - 1) / T;
            k_logsoftmax<<<blocks, T, 0, stream>>>(hbuf, outp, N);
        }
    }
}
